// SwitchMoE_28656021799161
// MI455X (gfx1250) — compile-verified, matrix operand order checked by reading, not run on hardware
//
#include <hip/hip_runtime.h>
#include <math.h>

// ---------------------------------------------------------------------------
// Switch-MoE (top-1) forward for MI455X / gfx1250, bf16 WMMA compute path.
// Double-buffered LDS pipeline, one barrier per K-step, packed B staging.
// ---------------------------------------------------------------------------

typedef __bf16 v16bf __attribute__((ext_vector_type(16)));
typedef float  v8f   __attribute__((ext_vector_type(8)));

#define D_MODEL 1024
#define D_FF    4096
#define N_EXP   8
#define N_TOK   8192
#define BM 128
#define BN 128
#define BK 32
#define LDSS 40   // LDS row stride in bf16 elements (80B: 16B aligned, padded)

__device__ __forceinline__ unsigned short f2bf(float f) {
  unsigned int u = __float_as_uint(f);
  u += 0x7FFFu + ((u >> 16) & 1u);      // round-to-nearest-even
  return (unsigned short)(u >> 16);
}

union Frag16 {
  uint4 u[2];
  v16bf v;
};

// --------------------------- x fp32 -> bf16 --------------------------------
__global__ __launch_bounds__(256) void moe_cvt_x_kernel(
    const float* __restrict__ x, unsigned short* __restrict__ xbf) {
  const int i = (blockIdx.x * 256 + threadIdx.x) * 4;
  const float4 f = *(const float4*)(x + i);
  uint2 p;
  p.x = (unsigned int)f2bf(f.x) | ((unsigned int)f2bf(f.y) << 16);
  p.y = (unsigned int)f2bf(f.z) | ((unsigned int)f2bf(f.w) << 16);
  *(uint2*)(xbf + i) = p;
}

// ------------------------------- router ------------------------------------
// One wave32 per token: lane-parallel dot over D, shuffle reduce, softmax,
// argmax (first-index wins on ties, matching jnp.argmax), atomic list build.
__global__ __launch_bounds__(256) void moe_router_kernel(
    const float* __restrict__ x, const float* __restrict__ rw,
    float* __restrict__ prob, int* __restrict__ lists, int* __restrict__ counts) {
  const int lane = threadIdx.x & 31;
  const int wid  = threadIdx.x >> 5;
  const int tok  = blockIdx.x * 8 + wid;
  const float* xr = x + (size_t)tok * D_MODEL;

  float acc[N_EXP];
#pragma unroll
  for (int e = 0; e < N_EXP; ++e) acc[e] = 0.0f;

  for (int d = lane; d < D_MODEL; d += 32) {
    const float xv = xr[d];
    const float4* r4 = (const float4*)(rw + (size_t)d * N_EXP);
    const float4 r0 = r4[0], r1 = r4[1];
    acc[0] += xv * r0.x; acc[1] += xv * r0.y;
    acc[2] += xv * r0.z; acc[3] += xv * r0.w;
    acc[4] += xv * r1.x; acc[5] += xv * r1.y;
    acc[6] += xv * r1.z; acc[7] += xv * r1.w;
  }
#pragma unroll
  for (int off = 16; off > 0; off >>= 1) {
#pragma unroll
    for (int e = 0; e < N_EXP; ++e) acc[e] += __shfl_xor(acc[e], off, 32);
  }
  if (lane == 0) {
    float mx = acc[0];
    int ei = 0;
#pragma unroll
    for (int e = 1; e < N_EXP; ++e) {
      if (acc[e] > mx) { mx = acc[e]; ei = e; }
    }
    float s = 0.0f;
#pragma unroll
    for (int e = 0; e < N_EXP; ++e) s += expf(acc[e] - mx);
    prob[tok] = 1.0f / s;                       // softmax prob of the argmax
    const int rank = atomicAdd(&counts[ei], 1);
    lists[ei * N_TOK + rank] = tok;
  }
}

// ------------------- fused expert GEMM (both FFN stages) -------------------
// STAGE1: H = gelu(X W1 + b1)   (bf16 out, scattered by token id)
// STAGE2: Y = (H W2 + b2) * p   (fp32 out)
template <int KLEN, int NWIDTH, bool STAGE1>
__global__ __launch_bounds__(256) void moe_ffn_kernel(
    const unsigned short* __restrict__ a_src,  // bf16 activations, row stride KLEN
    const float* __restrict__ w,               // (N_EXP, KLEN, NWIDTH) fp32
    const float* __restrict__ bias,            // (N_EXP, NWIDTH)
    const int* __restrict__ lists, const int* __restrict__ counts,
    const float* __restrict__ prob,            // stage2 only
    unsigned short* __restrict__ h_out,        // stage1 only
    float* __restrict__ y_out) {               // stage2 only
  const int e   = blockIdx.z;
  const int cnt = counts[e];
  const int m0  = blockIdx.y * BM;
  if (m0 >= cnt) return;
  const int n0  = blockIdx.x * BN;
  const int* list = lists + e * N_TOK;

  __shared__ unsigned short As[2][BM * LDSS];
  __shared__ unsigned short Bs[2][BN * LDSS];

  const int tid  = threadIdx.x;
  const int lane = tid & 31;
  const int wid  = tid >> 5;
  const int wm   = (wid & 3) * 32;   // wave's M offset in tile
  const int wn   = (wid >> 2) * 64;  // wave's N offset in tile

  // A staging: 2 threads per row, 16 bf16 each (two 16B loads/stores)
  const int lrow = tid >> 1;
  const int lseg = (tid & 1) * 16;
  int arow = m0 + lrow; if (arow > cnt - 1) arow = cnt - 1;
  const unsigned short* aptr = a_src + (size_t)list[arow] * KLEN + lseg;

  // B staging: thread covers 2 k-rows x 8 n; packs (k,k+1) pairs -> ds_store_b32
  const int n0b = (tid & 15) * 8;
  const int kp  = tid >> 4;          // 0..15 -> k = 2*kp
  const float* wrow =
      w + (size_t)e * KLEN * NWIDTH + (size_t)(2 * kp) * NWIDTH + n0 + n0b;

  uint4 a0, a1;
  float4 r00, r01, r10, r11;

  auto load_tile = [&](int k0) {
    a0 = *(const uint4*)(aptr + k0);
    a1 = *(const uint4*)(aptr + k0 + 8);
    const float* wr = wrow + (size_t)k0 * NWIDTH;
    r00 = ((const float4*)wr)[0];
    r01 = ((const float4*)wr)[1];
    const float* wr2 = wr + NWIDTH;
    r10 = ((const float4*)wr2)[0];
    r11 = ((const float4*)wr2)[1];
    __builtin_prefetch(wr + (size_t)BK * NWIDTH, 0, 1);
  };

  auto stage_store = [&](int buf) {
    *(uint4*)(&As[buf][lrow * LDSS + lseg])     = a0;
    *(uint4*)(&As[buf][lrow * LDSS + lseg + 8]) = a1;
    const float lo[8] = {r00.x, r00.y, r00.z, r00.w, r01.x, r01.y, r01.z, r01.w};
    const float hi[8] = {r10.x, r10.y, r10.z, r10.w, r11.x, r11.y, r11.z, r11.w};
#pragma unroll
    for (int j = 0; j < 8; ++j) {
      const unsigned int p =
          (unsigned int)f2bf(lo[j]) | ((unsigned int)f2bf(hi[j]) << 16);
      *(unsigned int*)(&Bs[buf][(n0b + j) * LDSS + 2 * kp]) = p;
    }
  };

  v8f c[2][4];
#pragma unroll
  for (int i = 0; i < 2; ++i)
#pragma unroll
    for (int j = 0; j < 4; ++j)
      c[i][j] = (v8f){0.f, 0.f, 0.f, 0.f, 0.f, 0.f, 0.f, 0.f};

  load_tile(0);
  stage_store(0);
  __syncthreads();

  for (int k0 = 0; k0 < KLEN; k0 += BK) {
    const int cur = (k0 >> 5) & 1;
    const bool more = (k0 + BK) < KLEN;
    if (more) load_tile(k0 + BK);   // global loads overlap this tile's WMMAs

    // CDNA5 bf16 fragment layout: lanes 0-15 hold K 0-7 & 16-23,
    // lanes 16-31 hold K 8-15 & 24-31 (two 16B LDS loads per fragment).
    const int koff = (lane & 16) ? 8 : 0;
    Frag16 fa[2], fb[4];
#pragma unroll
    for (int sm = 0; sm < 2; ++sm) {
      const unsigned short* p = &As[cur][(wm + sm * 16 + (lane & 15)) * LDSS + koff];
      fa[sm].u[0] = *(const uint4*)(p);
      fa[sm].u[1] = *(const uint4*)(p + 16);
    }
#pragma unroll
    for (int sn = 0; sn < 4; ++sn) {
      const unsigned short* p = &Bs[cur][(wn + sn * 16 + (lane & 15)) * LDSS + koff];
      fb[sn].u[0] = *(const uint4*)(p);
      fb[sn].u[1] = *(const uint4*)(p + 16);
    }
#pragma unroll
    for (int sm = 0; sm < 2; ++sm)
#pragma unroll
      for (int sn = 0; sn < 4; ++sn)
        c[sm][sn] = __builtin_amdgcn_wmma_f32_16x16x32_bf16(
            false, fa[sm].v, false, fb[sn].v, (short)0, c[sm][sn], false, false);

    if (more) stage_store(cur ^ 1);
    __syncthreads();
  }

  // ------------------------------ epilogue ---------------------------------
  const int rowhi = (lane >> 4) * 8;
  const int ncl   = lane & 15;
#pragma unroll
  for (int sm = 0; sm < 2; ++sm) {
#pragma unroll
    for (int sn = 0; sn < 4; ++sn) {
      const int col = n0 + wn + sn * 16 + ncl;
      const float bv = bias[(size_t)e * NWIDTH + col];
#pragma unroll
      for (int r = 0; r < 8; ++r) {
        const int gm = m0 + wm + sm * 16 + rowhi + r;
        if (gm < cnt) {
          const int tok = list[gm];
          if constexpr (STAGE1) {
            float v = c[sm][sn][r] + bv;
            v = 0.5f * v * (1.0f + erff(v * 0.70710678118654752f));
            h_out[(size_t)tok * NWIDTH + col] = f2bf(v);
          } else {
            y_out[(size_t)tok * NWIDTH + col] =
                (c[sm][sn][r] + bv) * prob[tok];
          }
        }
      }
    }
  }
}

// ------------------------------- launcher ----------------------------------
extern "C" void kernel_launch(void* const* d_in, const int* in_sizes, int n_in,
                              void* d_out, int out_size, void* d_ws, size_t ws_size,
                              hipStream_t stream) {
  (void)in_sizes; (void)n_in; (void)out_size; (void)ws_size;
  const float* x   = (const float*)d_in[0];  // (8192, 1024)
  const float* rw  = (const float*)d_in[1];  // (1024, 8)
  const float* w1  = (const float*)d_in[2];  // (8, 1024, 4096)
  const float* b1  = (const float*)d_in[3];  // (8, 4096)
  const float* w2  = (const float*)d_in[4];  // (8, 4096, 1024)
  const float* b2  = (const float*)d_in[5];  // (8, 1024)
  float* out = (float*)d_out;

  // Workspace layout (~84.2 MB total)
  char* ws = (char*)d_ws;
  unsigned short* xbf = (unsigned short*)ws;                       // 16 MiB
  unsigned short* h   = (unsigned short*)(ws + 16777216);          // 64 MiB
  float* prob         = (float*)(ws + 16777216 + 67108864);        // 32 KiB
  int*   lists        = (int*)(ws + 16777216 + 67108864 + 32768);  // 256 KiB
  int*   counts       = (int*)(ws + 16777216 + 67108864 + 32768 + 262144); // 64 B

  hipMemsetAsync(counts, 0, 64, stream);

  moe_cvt_x_kernel<<<(N_TOK * D_MODEL) / (256 * 4), 256, 0, stream>>>(x, xbf);
  moe_router_kernel<<<N_TOK / 8, 256, 0, stream>>>(x, rw, prob, lists, counts);
  moe_ffn_kernel<D_MODEL, D_FF, true>
      <<<dim3(D_FF / BN, N_TOK / BM, N_EXP), 256, 0, stream>>>(
          xbf, w1, b1, lists, counts, nullptr, h, nullptr);
  moe_ffn_kernel<D_FF, D_MODEL, false>
      <<<dim3(D_MODEL / BN, N_TOK / BM, N_EXP), 256, 0, stream>>>(
          h, w2, b2, lists, counts, prob, nullptr, out);
}
